// MSAModule_53300544143973
// MI455X (gfx1250) — compile-verified, matrix operand order checked by reading, not run on hardware
//
#include <hip/hip_runtime.h>
#include <hip/hip_bf16.h>

typedef __attribute__((ext_vector_type(16))) _Float16 v16h;
typedef __attribute__((ext_vector_type(8)))  _Float16 v8h;
typedef __attribute__((ext_vector_type(8)))  float    v8f;

// ---------------- dims ----------------
constexpr int S_MSA = 128;   // N_MSA
constexpr int T_TOK = 256;   // N_TOK
constexpr int CZ    = 128;
constexpr int CM    = 64;
constexpr int CS    = 384;
constexpr long long MT  = (long long)S_MSA * T_TOK;  // 32768 msa rows
constexpr long long TT2 = (long long)T_TOK * T_TOK;  // 65536 z rows

// WMMA 16x16x32 f16 fragment K-mapping: element e of lane-half -> K offset.
__device__ __forceinline__ int kmap16(int e, int half) {
  return (e < 8) ? (half * 8 + e) : (16 + half * 8 + (e - 8));
}

__device__ __forceinline__ float fast_rcp(float x) { return __builtin_amdgcn_rcpf(x); }
__device__ __forceinline__ float sigf(float x) { return fast_rcp(1.f + __expf(-x)); }
__device__ __forceinline__ float4 f4mul(float4 a, float4 b) {
  return make_float4(a.x * b.x, a.y * b.y, a.z * b.z, a.w * b.w);
}

// ---------------- GEMM descriptor ----------------
struct GemmP {
  const float* A; const float* B; float* C;
  const float* bias;   // per output column, or null
  const float* amul;   // elementwise multiplier on A (same strides as A), or null
  const float* cmul;   // elementwise multiplier on result (same strides as C), or null
  int M, N, K, B1, B2;
  long long sAm, sAk, sA1, sA2;
  long long sBk, sBn, sB1, sB2;
  long long sCm, sCn, sC1, sC2;
  float alpha;
  int epi;    // 0 none, 1 sigmoid
  int accum;  // 0 store, 1 +=
};

// vectorized contiguous A fragment load (global): two 8-float runs -> 4x b128
__device__ __forceinline__ v16h load_a_vec(const float* Arow, int k0, int half) {
  const float4* p0 = (const float4*)(Arow + k0 + half * 8);
  const float4* p1 = (const float4*)(Arow + k0 + 16 + half * 8);
  float4 x0 = p0[0], x1 = p0[1], y0 = p1[0], y1 = p1[1];
  v16h a;
  a[0] = (_Float16)x0.x; a[1] = (_Float16)x0.y; a[2] = (_Float16)x0.z; a[3] = (_Float16)x0.w;
  a[4] = (_Float16)x1.x; a[5] = (_Float16)x1.y; a[6] = (_Float16)x1.z; a[7] = (_Float16)x1.w;
  a[8] = (_Float16)y0.x; a[9] = (_Float16)y0.y; a[10] = (_Float16)y0.z; a[11] = (_Float16)y0.w;
  a[12] = (_Float16)y1.x; a[13] = (_Float16)y1.y; a[14] = (_Float16)y1.z; a[15] = (_Float16)y1.w;
  return a;
}

// =====================================================================
// LDS-tiled GEMM: block = 128 threads (4 waves), 64x64 output tile.
// Requires M%64==0, N%64==0, K%32==0, sAk==1, and (BK_CONT ? sBk==1 : sBn==1).
// =====================================================================
template <bool HAS_AM, bool BK_CONT>
__global__ __launch_bounds__(128) void gemm_tile_kernel(GemmP p) {
  __shared__ _Float16 At[64][32];
  __shared__ _Float16 Bt[64][32];   // stored [col][k]
  const int n0 = blockIdx.x * 64;
  const int m0 = blockIdx.y * 64;
  const int bz = blockIdx.z;
  const int b2 = bz % p.B2, b1 = bz / p.B2;
  const float* __restrict__ A = p.A + b1 * p.sA1 + b2 * p.sA2;
  const float* __restrict__ B = p.B + b1 * p.sB1 + b2 * p.sB2;
  float* __restrict__ C = p.C + b1 * p.sC1 + b2 * p.sC2;
  const float* AMb = HAS_AM ? p.amul + b1 * p.sA1 + b2 * p.sA2 : nullptr;
  const float* CMl = p.cmul ? p.cmul + b1 * p.sC1 + b2 * p.sC2 : nullptr;
  const int tid = threadIdx.x;
  const int w = tid >> 5, lane = tid & 31, half = lane >> 4, r = lane & 15;
  const int srow = tid >> 1;           // 0..63
  const int skh = (tid & 1) * 16;      // 0 or 16

  v8f acc[4] = {{}, {}, {}, {}};
  for (int k0 = 0; k0 < p.K; k0 += 32) {
    __syncthreads();
    // ---- stage A tile (64 rows x 32 k), f32 -> f16 ----
    {
      const float* Ar = A + (long long)(m0 + srow) * p.sAm + (k0 + skh);
      const float4* Av = (const float4*)Ar;
      float4 f0 = Av[0], f1 = Av[1], f2 = Av[2], f3 = Av[3];
      if (HAS_AM) {
        const float4* Mv = (const float4*)(AMb + (long long)(m0 + srow) * p.sAm + (k0 + skh));
        f0 = f4mul(f0, Mv[0]); f1 = f4mul(f1, Mv[1]);
        f2 = f4mul(f2, Mv[2]); f3 = f4mul(f3, Mv[3]);
      }
      float ff[16] = {f0.x, f0.y, f0.z, f0.w, f1.x, f1.y, f1.z, f1.w,
                      f2.x, f2.y, f2.z, f2.w, f3.x, f3.y, f3.z, f3.w};
      _Float16* dst = &At[srow][skh];
#pragma unroll
      for (int u = 0; u < 16; ++u) dst[u] = (_Float16)ff[u];
    }
    // ---- stage B tile into Bt[col][k] ----
    if (BK_CONT) {
      const float* Br = B + (long long)(n0 + srow) * p.sBn + (k0 + skh);
      const float4* Bv = (const float4*)Br;
      float4 f0 = Bv[0], f1 = Bv[1], f2 = Bv[2], f3 = Bv[3];
      float ff[16] = {f0.x, f0.y, f0.z, f0.w, f1.x, f1.y, f1.z, f1.w,
                      f2.x, f2.y, f2.z, f2.w, f3.x, f3.y, f3.z, f3.w};
      _Float16* dst = &Bt[srow][skh];
#pragma unroll
      for (int u = 0; u < 16; ++u) dst[u] = (_Float16)ff[u];
    } else {
      int krow = tid >> 2;             // 0..31
      int nseg = (tid & 3) * 16;       // 0,16,32,48
      const float* Br = B + (long long)(k0 + krow) * p.sBk + (n0 + nseg);
      const float4* Bv = (const float4*)Br;
      float4 f0 = Bv[0], f1 = Bv[1], f2 = Bv[2], f3 = Bv[3];
      float ff[16] = {f0.x, f0.y, f0.z, f0.w, f1.x, f1.y, f1.z, f1.w,
                      f2.x, f2.y, f2.z, f2.w, f3.x, f3.y, f3.z, f3.w};
#pragma unroll
      for (int u = 0; u < 16; ++u) Bt[nseg + u][krow] = (_Float16)ff[u];
    }
    __syncthreads();
    // ---- compute: B frag shared across 4 M-tiles, b128 LDS reads ----
    v16h kb;
    {
      v8h lo = *(const v8h*)&Bt[w * 16 + r][half * 8];
      v8h hi = *(const v8h*)&Bt[w * 16 + r][16 + half * 8];
#pragma unroll
      for (int e = 0; e < 8; ++e) { kb[e] = lo[e]; kb[8 + e] = hi[e]; }
    }
#pragma unroll
    for (int t = 0; t < 4; ++t) {
      v8h lo = *(const v8h*)&At[t * 16 + r][half * 8];
      v8h hi = *(const v8h*)&At[t * 16 + r][16 + half * 8];
      v16h af;
#pragma unroll
      for (int e = 0; e < 8; ++e) { af[e] = lo[e]; af[8 + e] = hi[e]; }
      acc[t] = __builtin_amdgcn_wmma_f32_16x16x32_f16(false, af, false, kb, (short)0, acc[t], false, false);
    }
  }
  const int ncol = n0 + w * 16 + r;
#pragma unroll
  for (int t = 0; t < 4; ++t) {
#pragma unroll
    for (int i = 0; i < 8; ++i) {
      int row = m0 + t * 16 + i + 8 * half;
      float v = p.alpha * acc[t][i];
      if (p.bias) v += p.bias[ncol];
      if (p.epi == 1) v = sigf(v);
      long long ic = (long long)row * p.sCm + (long long)ncol * p.sCn;
      if (CMl) v *= CMl[ic];
      if (p.accum) C[ic] += v; else C[ic] = v;
    }
  }
}

// Strided / guarded fallback. CHECK=false assumes M%16==0, N%16==0, K%32==0.
template <bool CHECK>
__global__ __launch_bounds__(32) void gemm_kernel(GemmP p) {
  const int n0 = blockIdx.x * 16;
  const int m0 = blockIdx.y * 16;
  const int bz = blockIdx.z;
  const int b2 = bz % p.B2, b1 = bz / p.B2;
  const float* __restrict__ A = p.A + b1 * p.sA1 + b2 * p.sA2;
  const float* __restrict__ B = p.B + b1 * p.sB1 + b2 * p.sB2;
  float* __restrict__ C = p.C + b1 * p.sC1 + b2 * p.sC2;
  const bool hasAM = p.amul != nullptr;
  const float* AMb = hasAM ? p.amul + b1 * p.sA1 + b2 * p.sA2 : nullptr;
  const float* CMl = p.cmul ? p.cmul + b1 * p.sC1 + b2 * p.sC2 : nullptr;
  const int lane = threadIdx.x & 31, half = lane >> 4, r = lane & 15;
  const int mrow = m0 + r;
  const int ncol = n0 + r;
  const bool mok = !CHECK || (mrow < p.M);
  const bool nok = !CHECK || (ncol < p.N);
  v8f acc = {};
  for (int k0 = 0; k0 < p.K; k0 += 32) {
    v16h a = {}, b = {};
#pragma unroll
    for (int e = 0; e < 16; ++e) {
      int kk = k0 + kmap16(e, half);
      float av = 0.f, bv = 0.f;
      if (mok && (!CHECK || kk < p.K)) {
        long long ia = (long long)mrow * p.sAm + (long long)kk * p.sAk;
        av = A[ia];
        if (hasAM) av *= AMb[ia];
      }
      if (nok && (!CHECK || kk < p.K))
        bv = B[(long long)kk * p.sBk + (long long)ncol * p.sBn];
      a[e] = (_Float16)av;
      b[e] = (_Float16)bv;
    }
    acc = __builtin_amdgcn_wmma_f32_16x16x32_f16(false, a, false, b, (short)0, acc, false, false);
  }
#pragma unroll
  for (int i = 0; i < 8; ++i) {
    int row = m0 + i + 8 * half;
    int col = n0 + r;
    if (!CHECK || (row < p.M && col < p.N)) {
      float v = p.alpha * acc[i];
      if (p.bias) v += p.bias[col];
      if (p.epi == 1) v = sigf(v);
      long long ic = (long long)row * p.sCm + (long long)col * p.sCn;
      if (CMl) v *= CMl[ic];
      if (p.accum) C[ic] += v; else C[ic] = v;
    }
  }
}

// OUT = act(A@W1) * (A@W2). Assumes M%16==0, N%16==0, K%32==0, row-major.
__global__ __launch_bounds__(32) void dual_gemm_kernel(
    const float* __restrict__ A, const float* __restrict__ W1,
    const float* __restrict__ W2, float* __restrict__ OUT,
    int M, int N, int K, int act /*1 sigmoid, 2 silu*/) {
  const int n0 = blockIdx.x * 16, m0 = blockIdx.y * 16;
  const int lane = threadIdx.x & 31, half = lane >> 4, r = lane & 15;
  const int ncol = n0 + r;
  const float* Arow = A + (long long)(m0 + r) * K;
  v8f acc1 = {}, acc2 = {};
  for (int k0 = 0; k0 < K; k0 += 32) {
    v16h af = load_a_vec(Arow, k0, half);
    v16h b1, b2;
    const float* W1p = W1 + (long long)(k0 + half * 8) * N + ncol;
    const float* W2p = W2 + (long long)(k0 + half * 8) * N + ncol;
#pragma unroll
    for (int e = 0; e < 8; ++e) {
      b1[e] = (_Float16)W1p[(long long)e * N];
      b1[8 + e] = (_Float16)W1p[(long long)(16 + e) * N];
      b2[e] = (_Float16)W2p[(long long)e * N];
      b2[8 + e] = (_Float16)W2p[(long long)(16 + e) * N];
    }
    acc1 = __builtin_amdgcn_wmma_f32_16x16x32_f16(false, af, false, b1, (short)0, acc1, false, false);
    acc2 = __builtin_amdgcn_wmma_f32_16x16x32_f16(false, af, false, b2, (short)0, acc2, false, false);
  }
#pragma unroll
  for (int i = 0; i < 8; ++i) {
    int row = m0 + i + 8 * half, col = n0 + r;
    float g = acc1[i];
    g = (act == 1) ? sigf(g) : g * sigf(g);
    OUT[(long long)row * N + col] = g * acc2[i];
  }
}

// channel transpose for tri-mul: abt[c][x*256+y] = ab[(row)*256+c],
// row = swap ? y*256+x : x*256+y ; c in [0,256)
__global__ void chan_transpose_kernel(const float* __restrict__ ab,
                                      float* __restrict__ abt, int swap) {
  long long idx = (long long)blockIdx.x * blockDim.x + threadIdx.x;
  if (idx >= 256LL * TT2) return;
  int y = (int)(idx & 255);
  long long rem = idx >> 8;
  int x = (int)(rem & 255);
  long long c = rem >> 8;
  long long row = swap ? ((long long)y * 256 + x) : ((long long)x * 256 + y);
  abt[idx] = ab[row * 256 + c];
}

// ---------------- LayerNorm: one wave per row ----------------
__global__ void ln_kernel(const float* __restrict__ x, float* __restrict__ y,
                          long long rows, int C,
                          const float* __restrict__ g, const float* __restrict__ b) {
  long long row = (long long)blockIdx.x * (blockDim.x >> 5) + (threadIdx.x >> 5);
  if (row >= rows) return;
  int lane = threadIdx.x & 31;
  int per = C >> 5;  // C in {64,128}
  const float* xr = x + row * C;
  float v[4];
  float s = 0.f;
  for (int i = 0; i < per; ++i) { v[i] = xr[i * 32 + lane]; s += v[i]; }
  for (int off = 16; off; off >>= 1) s += __shfl_xor(s, off, 32);
  float mu = s / C;
  float q = 0.f;
  for (int i = 0; i < per; ++i) { float d = v[i] - mu; q += d * d; }
  for (int off = 16; off; off >>= 1) q += __shfl_xor(q, off, 32);
  float rstd = rsqrtf(q / C + 1e-5f);
  float* yr = y + row * C;
  for (int i = 0; i < per; ++i) {
    int c = i * 32 + lane;
    yr[c] = (v[i] - mu) * rstd * g[c] + b[c];
  }
}

// ---------------- input embedding ----------------
__global__ void embed_kernel(const float* __restrict__ msa, const float* __restrict__ hd,
                             const float* __restrict__ dv, const float* __restrict__ lin1,
                             const float* __restrict__ s2, float* __restrict__ m) {
  long long idx = (long long)blockIdx.x * blockDim.x + threadIdx.x;
  if (idx >= MT * CM) return;
  int c = idx & 63;
  long long st = idx >> 6;
  int t = (int)(st & 255);
  float acc = s2[t * CM + c];
  const float* mr = msa + st * 32;
#pragma unroll
  for (int f = 0; f < 32; ++f) acc += mr[f] * lin1[f * CM + c];
  acc += hd[st] * lin1[32 * CM + c] + dv[st] * lin1[33 * CM + c];
  m[idx] = acc;
}

// ---------------- pwa column softmax over i, per (j,h), 8 heads ----------------
__global__ void pwa_softmax_kernel(const float* __restrict__ bias, float* __restrict__ w) {
  int idx = blockIdx.x * blockDim.x + threadIdx.x;
  if (idx >= T_TOK * 8) return;
  int j = idx >> 3, h = idx & 7;
  float mx = -1e30f;
  for (int i = 0; i < T_TOK; ++i)
    mx = fmaxf(mx, bias[((long long)i * T_TOK + j) * 8 + h]);
  float s = 0.f;
  for (int i = 0; i < T_TOK; ++i) {
    float e = __expf(bias[((long long)i * T_TOK + j) * 8 + h] - mx);
    w[((long long)i * T_TOK + j) * 8 + h] = e;
    s += e;
  }
  float inv = fast_rcp(s);
  for (int i = 0; i < T_TOK; ++i)
    w[((long long)i * T_TOK + j) * 8 + h] *= inv;
}

// z[i,j,c] += t[j,c]
__global__ void bcast_add_z_kernel(float* __restrict__ z, const float* __restrict__ t) {
  long long idx = (long long)blockIdx.x * blockDim.x + threadIdx.x;
  if (idx >= TT2 * CZ) return;
  z[idx] += t[idx % ((long long)T_TOK * CZ)];
}

// ---------------- fused triangle attention (per block: one (i, head)) ----------------
__global__ __launch_bounds__(128) void tri_attn_kernel(
    const float* __restrict__ qkv, const float* __restrict__ bias,
    float* __restrict__ o, int starting) {
  __shared__ _Float16 ks[256][32];
  __shared__ _Float16 vs[256][32];
  __shared__ _Float16 ps[4][16][256];
  const int i = blockIdx.x;
  const int h = blockIdx.y;
  const int tid = threadIdx.x;
  const int wave = tid >> 5;
  const int lane = tid & 31;
  const int half = lane >> 4, r = lane & 15;

  for (int idx = tid; idx < 256 * 32; idx += 128) {
    int k = idx >> 5, c = idx & 31;
    long long row = starting ? ((long long)i * 256 + k) : ((long long)k * 256 + i);
    const float* qr = qkv + row * 384;
    ks[k][c] = (_Float16)qr[128 + h * 32 + c];
    vs[k][c] = (_Float16)qr[256 + h * 32 + c];
  }
  __syncthreads();

  const float scale = 0.1767766952966369f;  // 1/sqrt(32)
  for (int t = 0; t < 4; ++t) {
    int j0 = wave * 64 + t * 16;
    v16h qa = load_a_vec(qkv + ((long long)i * 256 + (j0 + r)) * 384 + h * 32, 0, half);
    v8f sfr[16];
#pragma unroll
    for (int kc = 0; kc < 16; ++kc) {
      v16h kb;
#pragma unroll
      for (int e = 0; e < 16; ++e) kb[e] = ks[kc * 16 + r][kmap16(e, half)];
      v8f acc = {};
      acc = __builtin_amdgcn_wmma_f32_16x16x32_f16(false, qa, false, kb, (short)0, acc, false, false);
      int kcol = kc * 16 + r;
#pragma unroll
      for (int e = 0; e < 8; ++e) {
        int j = j0 + e + 8 * half;
        float bv = starting ? bias[((long long)j * 256 + kcol) * 4 + h]
                            : bias[((long long)kcol * 256 + i) * 4 + h];
        acc[e] = scale * acc[e] + bv;
      }
      sfr[kc] = acc;
    }
    float rsum[8];
#pragma unroll
    for (int e = 0; e < 8; ++e) {
      float mx = -1e30f;
#pragma unroll
      for (int kc = 0; kc < 16; ++kc) mx = fmaxf(mx, sfr[kc][e]);
      mx = fmaxf(mx, __shfl_xor(mx, 1, 32));
      mx = fmaxf(mx, __shfl_xor(mx, 2, 32));
      mx = fmaxf(mx, __shfl_xor(mx, 4, 32));
      mx = fmaxf(mx, __shfl_xor(mx, 8, 32));
      float s = 0.f;
#pragma unroll
      for (int kc = 0; kc < 16; ++kc) {
        float pv = __expf(sfr[kc][e] - mx);
        sfr[kc][e] = pv;
        s += pv;
      }
      s += __shfl_xor(s, 1, 32);
      s += __shfl_xor(s, 2, 32);
      s += __shfl_xor(s, 4, 32);
      s += __shfl_xor(s, 8, 32);
      rsum[e] = s;
    }
#pragma unroll
    for (int kc = 0; kc < 16; ++kc) {
      int kcol = kc * 16 + r;
#pragma unroll
      for (int e = 0; e < 8; ++e)
        ps[wave][e + 8 * half][kcol] = (_Float16)sfr[kc][e];
    }
    __syncthreads();
    v8f o0 = {}, o1 = {};
#pragma unroll
    for (int kc = 0; kc < 8; ++kc) {
      v16h pa, v0, v1;
      {
        v8h lo = *(const v8h*)&ps[wave][r][kc * 32 + half * 8];
        v8h hi = *(const v8h*)&ps[wave][r][kc * 32 + 16 + half * 8];
#pragma unroll
        for (int e = 0; e < 8; ++e) { pa[e] = lo[e]; pa[8 + e] = hi[e]; }
      }
#pragma unroll
      for (int e = 0; e < 16; ++e) {
        int kk = kc * 32 + kmap16(e, half);
        v0[e] = vs[kk][r];
        v1[e] = vs[kk][16 + r];
      }
      o0 = __builtin_amdgcn_wmma_f32_16x16x32_f16(false, pa, false, v0, (short)0, o0, false, false);
      o1 = __builtin_amdgcn_wmma_f32_16x16x32_f16(false, pa, false, v1, (short)0, o1, false, false);
    }
#pragma unroll
    for (int e = 0; e < 8; ++e) {
      int j = j0 + e + 8 * half;
      float inv = fast_rcp(rsum[e]);
      long long orow = ((long long)i * 256 + j) * 128 + h * 32;
      o[orow + r] = o0[e] * inv;
      o[orow + 16 + r] = o1[e] * inv;
    }
    __syncthreads();
  }
}

// ---------------- host-side helpers ----------------
static GemmP mk(const float* A, const float* B, float* C, int M, int N, int K) {
  GemmP p{};
  p.A = A; p.B = B; p.C = C;
  p.M = M; p.N = N; p.K = K; p.B1 = 1; p.B2 = 1;
  p.sAm = K; p.sAk = 1; p.sBk = N; p.sBn = 1; p.sCm = N; p.sCn = 1;
  p.alpha = 1.f;
  return p;
}
static void launch_gemm(hipStream_t s, const GemmP& p) {
  const bool tile_ok = (p.M % 64 == 0) && (p.N % 64 == 0) && (p.K % 32 == 0) &&
                       (p.sAk == 1) && (p.sBn == 1 || p.sBk == 1);
  if (tile_ok) {
    dim3 grid(p.N / 64, p.M / 64, p.B1 * p.B2);
    if (p.sBk == 1) {
      if (p.amul) hipLaunchKernelGGL((gemm_tile_kernel<true, true>), grid, dim3(128), 0, s, p);
      else        hipLaunchKernelGGL((gemm_tile_kernel<false, true>), grid, dim3(128), 0, s, p);
    } else {
      if (p.amul) hipLaunchKernelGGL((gemm_tile_kernel<true, false>), grid, dim3(128), 0, s, p);
      else        hipLaunchKernelGGL((gemm_tile_kernel<false, false>), grid, dim3(128), 0, s, p);
    }
  } else if ((p.M % 16 == 0) && (p.N % 16 == 0) && (p.K % 32 == 0)) {
    dim3 grid(p.N / 16, p.M / 16, p.B1 * p.B2);
    hipLaunchKernelGGL(gemm_kernel<false>, grid, dim3(32), 0, s, p);
  } else {
    dim3 grid((p.N + 15) / 16, (p.M + 15) / 16, p.B1 * p.B2);
    hipLaunchKernelGGL(gemm_kernel<true>, grid, dim3(32), 0, s, p);
  }
}
static void launch_dual(hipStream_t s, const float* A, const float* W1, const float* W2,
                        float* OUT, int M, int N, int K, int act) {
  dim3 grid(N / 16, M / 16, 1);
  hipLaunchKernelGGL(dual_gemm_kernel, grid, dim3(32), 0, s, A, W1, W2, OUT, M, N, K, act);
}
static void launch_ln(hipStream_t s, const float* x, float* y, long long rows, int C,
                      const float* g, const float* b) {
  dim3 grid((unsigned)((rows + 7) / 8));
  hipLaunchKernelGGL(ln_kernel, grid, dim3(256), 0, s, x, y, rows, C, g, b);
}

extern "C" void kernel_launch(void* const* d_in, const int* in_sizes, int n_in,
                              void* d_out, int out_size, void* d_ws, size_t ws_size,
                              hipStream_t stream) {
  (void)in_sizes; (void)n_in; (void)out_size; (void)ws_size;
  auto in = [&](int i) { return (const float*)d_in[i]; };
  float* z = (float*)d_out;

  float* ws = (float*)d_ws;
  size_t cur = 0;
  auto alloc = [&](size_t n) { float* p = ws + cur; cur += n; return p; };
  float* m    = alloc(MT * CM);
  float* mh   = alloc(MT * CM);
  float* s2   = alloc((size_t)T_TOK * CM);
  float* zh   = alloc(TT2 * CZ);
  float* hbig = alloc(TT2 * 512);       // transitions hidden / qkv / tri-mul ab (+abt)
  float* gs   = alloc(TT2 * CZ);
  float* prod = alloc(TT2 * CZ);
  float* pln  = alloc(TT2 * CZ);
  float* obuf = alloc(TT2 * CZ);
  float* bsm  = alloc(TT2 * 8);
  float* wsm  = alloc(TT2 * 8);
  float* oab  = alloc(MT * CM);
  float* oo   = alloc((size_t)T_TOK * 1024);
  float* of   = alloc((size_t)T_TOK * CZ);
  float* pv   = alloc(MT * CM);
  float* pg   = alloc(MT * CM);

  hipMemcpyAsync(z, d_in[3], TT2 * CZ * sizeof(float), hipMemcpyDeviceToDevice, stream);

  launch_gemm(stream, mk(in(4), in(6), s2, T_TOK, CM, CS));
  hipLaunchKernelGGL(embed_kernel, dim3((unsigned)((MT * CM + 255) / 256)), dim3(256), 0, stream,
                     in(0), in(1), in(2), in(5), s2, m);

  for (int blk = 0; blk < 2; ++blk) {
    int base = 7 + blk * 51;

    // ===== OPM: z += opm(m)[None] =====
    launch_ln(stream, m, mh, MT, CM, in(base + 0), in(base + 1));
    launch_gemm(stream, mk(mh, in(base + 2), oab, (int)MT, CM, CM));
    {
      GemmP p = mk(oab, oab + 32, oo, 32, 32, S_MSA);     // o[t,c,d] = mean_s a*b
      p.sAm = 1;  p.sAk = (long long)T_TOK * CM;
      p.sBk = (long long)T_TOK * CM; p.sBn = 1;
      p.sCm = 32; p.sCn = 1;
      p.B1 = T_TOK; p.sA1 = CM; p.sB1 = CM; p.sC1 = 1024;
      p.alpha = 1.f / S_MSA;
      launch_gemm(stream, p);
    }
    { GemmP p = mk(oo, in(base + 3), of, T_TOK, CZ, 1024); p.bias = in(base + 4); launch_gemm(stream, p); }
    hipLaunchKernelGGL(bcast_add_z_kernel, dim3((unsigned)((TT2 * CZ + 255) / 256)), dim3(256), 0, stream, z, of);

    // ===== PWA: m += pwa(m, z) =====
    launch_ln(stream, m, mh, MT, CM, in(base + 5), in(base + 6));
    launch_gemm(stream, mk(mh, in(base + 7), pv, (int)MT, CM, CM));
    { GemmP p = mk(mh, in(base + 11), pg, (int)MT, CM, CM); p.epi = 1; launch_gemm(stream, p); }
    launch_ln(stream, z, zh, TT2, CZ, in(base + 8), in(base + 9));
    launch_gemm(stream, mk(zh, in(base + 10), bsm, (int)TT2, 8, CZ));
    hipLaunchKernelGGL(pwa_softmax_kernel, dim3(8), dim3(256), 0, stream, bsm, wsm);
    {
      GemmP p = mk(wsm, pv, obuf, T_TOK, 8, T_TOK);       // o[s,i,h,c] = sum_j w[i,j,h] v[s,j,h,c]
      p.sAm = (long long)T_TOK * 8; p.sAk = 8;
      p.sBk = CM; p.sBn = 1;
      p.sCm = CM; p.sCn = 1;
      p.B1 = S_MSA; p.B2 = 8;
      p.sA1 = 0; p.sA2 = 1;
      p.sB1 = (long long)T_TOK * CM; p.sB2 = 8;
      p.sC1 = (long long)T_TOK * CM; p.sC2 = 8;
      p.cmul = pg;
      launch_gemm(stream, p);
    }
    { GemmP p = mk(obuf, in(base + 12), m, (int)MT, CM, CM); p.accum = 1; launch_gemm(stream, p); }

    // ===== TRM: m += transition(m) =====
    launch_ln(stream, m, mh, MT, CM, in(base + 13), in(base + 14));
    launch_dual(stream, mh, in(base + 15), in(base + 16), hbig, (int)MT, 256, CM, 2);
    { GemmP p = mk(hbig, in(base + 17), m, (int)MT, CM, 256); p.accum = 1; launch_gemm(stream, p); }

    // ===== TMO / TMI: z += tri_mul(z) =====
    for (int dir = 0; dir < 2; ++dir) {
      int tb = base + 18 + dir * 8;
      bool outgoing = (dir == 0);
      launch_ln(stream, z, zh, TT2, CZ, in(tb + 0), in(tb + 1));
      launch_dual(stream, zh, in(tb + 2), in(tb + 3), hbig, (int)TT2, 256, CZ, 1);
      { GemmP p = mk(zh, in(tb + 4), gs, (int)TT2, CZ, CZ); p.epi = 1; launch_gemm(stream, p); }
      float* abt = hbig + TT2 * 256;
      hipLaunchKernelGGL(chan_transpose_kernel,
                         dim3((unsigned)((256LL * TT2 + 255) / 256)), dim3(256), 0, stream,
                         hbig, abt, outgoing ? 0 : 1);
      {
        GemmP p = mk(abt, abt + (long long)128 * TT2, prod, T_TOK, T_TOK, T_TOK);
        p.sAm = 256; p.sAk = 1;
        p.sBk = 1;   p.sBn = 256;
        p.sCm = (long long)T_TOK * CZ; p.sCn = CZ;
        p.B1 = CZ; p.sA1 = TT2; p.sB1 = TT2; p.sC1 = 1;
        launch_gemm(stream, p);
      }
      launch_ln(stream, prod, pln, TT2, CZ, in(tb + 5), in(tb + 6));
      { GemmP p = mk(pln, in(tb + 7), z, (int)TT2, CZ, CZ); p.cmul = gs; p.accum = 1; launch_gemm(stream, p); }
    }

    // ===== TAS / TAE: z += tri_attn(z) =====
    for (int dir = 0; dir < 2; ++dir) {
      int tb = base + 34 + dir * 6;
      int starting = (dir == 0) ? 1 : 0;
      launch_ln(stream, z, zh, TT2, CZ, in(tb + 0), in(tb + 1));
      launch_gemm(stream, mk(zh, in(tb + 2), hbig, (int)TT2, 384, CZ));
      launch_gemm(stream, mk(zh, in(tb + 3), bsm, (int)TT2, 4, CZ));
      { GemmP p = mk(zh, in(tb + 4), gs, (int)TT2, CZ, CZ); p.epi = 1; launch_gemm(stream, p); }
      hipLaunchKernelGGL(tri_attn_kernel, dim3(T_TOK, 4), dim3(128), 0, stream,
                         hbig, bsm, obuf, starting);
      { GemmP p = mk(obuf, in(tb + 5), z, (int)TT2, CZ, CZ); p.amul = gs; p.accum = 1; launch_gemm(stream, p); }
    }

    // ===== TRZ: z += transition(z) =====
    int rb = base + 46;
    launch_ln(stream, z, zh, TT2, CZ, in(rb + 0), in(rb + 1));
    launch_dual(stream, zh, in(rb + 2), in(rb + 3), hbig, (int)TT2, 512, CZ, 2);
    { GemmP p = mk(hbig, in(rb + 4), z, (int)TT2, CZ, 512); p.accum = 1; launch_gemm(stream, p); }
  }
}
